// ScaledDotProductAttention_15788299780171
// MI455X (gfx1250) — matrix operand order repaired and checked by reading, not compiled, not run on hardware
//
#include <hip/hip_runtime.h>
#include <hip/hip_bf16.h>
#include <hip/hip_fp16.h>

typedef __attribute__((ext_vector_type(16))) _Float16 v16h;
typedef __attribute__((ext_vector_type(8)))  float    v8f;
typedef __attribute__((ext_vector_type(4)))  float    f32x4;

#define BSNH 32
#define NK   256
#define DK   64

// ---------------------------------------------------------------------------
// s_val = (sa * sv).mean(axis=2)  -> out[..., 64:128]
// Pure streaming: 1.07 GB read once -> nontemporal B128 loads, LDS tree reduce.
// One 256-thread block per (b,q): 16 k-partitions x 16 float4 columns.
// ---------------------------------------------------------------------------
__global__ __launch_bounds__(256) void sval_kernel(const float* __restrict__ sa,
                                                   const float* __restrict__ sv,
                                                   float* __restrict__ out)
{
    const int bq = blockIdx.x;           // b*NK + q
    const int t  = threadIdx.x;
    const int d4 = t & 15;               // float4 index within 64-float row
    const int kp = t >> 4;               // k partition 0..15
    const f32x4* sap = (const f32x4*)(sa + (size_t)bq * NK * DK);
    const f32x4* svp = (const f32x4*)(sv + (size_t)bq * NK * DK);

    f32x4 acc = {0.f, 0.f, 0.f, 0.f};
    for (int k = kp; k < NK; k += 16) {
        f32x4 a = __builtin_nontemporal_load(sap + k * 16 + d4);
        f32x4 b = __builtin_nontemporal_load(svp + k * 16 + d4);
        acc += a * b;
    }

    __shared__ f32x4 red[256];
    red[t] = acc;
    __syncthreads();
    for (int s = 8; s > 0; s >>= 1) {
        if (kp < s) red[t] += red[t + s * 16];
        __syncthreads();
    }
    if (kp == 0) {
        f32x4 r = red[t] * (1.0f / 256.0f);
        ((f32x4*)(out + (size_t)bq * 128 + 64))[d4] = r;
    }
}

// ---------------------------------------------------------------------------
// Attention: scores = QK^T/8, mask, softmax, P@V. One wave per 16-row q-tile.
// WMMA f32_16x16x32_f16. LDS 16x256 f32 score tile per wave, XOR-swizzled
// (8-aligned blocks) so strided A-fragment reads are bank-conflict light.
// ---------------------------------------------------------------------------
__global__ __launch_bounds__(128) void attn_kernel(const float* __restrict__ qg,
                                                   const float* __restrict__ kg,
                                                   const float* __restrict__ vg,
                                                   const int*   __restrict__ mg,
                                                   float* __restrict__ out,
                                                   float* __restrict__ attn)
{
    __shared__ float P[4][16 * 256];     // 64 KB total
    const int b    = blockIdx.x;
    const int wave = threadIdx.x >> 5;
    const int lane = threadIdx.x & 31;
    const int q0   = (blockIdx.y * 4 + wave) * 16;
    float* Pw = P[wave];

    const float* Qb = qg + ((size_t)b * NK + q0) * DK;
    const float* Kb = kg + (size_t)b * NK * DK;
    const float* Vb = vg + (size_t)b * NK * DK;
    const int*   Mb = mg + ((size_t)b * NK + q0) * NK;
    float* outb = out  + ((size_t)b * NK + q0) * 128;
    float* attb = attn + ((size_t)b * NK + q0) * NK;

    const int half = lane >> 4;          // 0: K 0..7 half, 1: K 8..15 half (A)
    const int l16  = lane & 15;
    const int aoff = half * 8;           // A-fragment K sub-offset
    const int boff = half * 8;           // B-fragment K sub-offset

    // ---- Q A-fragments: 16x32 f16, two chunks covering DK=64 ----
    v16h a0, a1;
    {
        const float* qrow = Qb + l16 * DK;
        #pragma unroll
        for (int j = 0; j < 8; ++j) {
            a0[j]     = (_Float16)qrow[ 0 + aoff + j];
            a0[8 + j] = (_Float16)qrow[16 + aoff + j];
            a1[j]     = (_Float16)qrow[32 + aoff + j];
            a1[8 + j] = (_Float16)qrow[48 + aoff + j];
        }
    }

    // ---- scores = QK^T / 8, mask, to swizzled LDS ----
    for (int kt = 0; kt < 16; ++kt) {
        v16h b0, b1;
        const float* krow = Kb + (size_t)(kt * 16 + l16) * DK;
        #pragma unroll
        for (int j = 0; j < 8; ++j) {
            b0[j]     = (_Float16)krow[ 0 + boff + j];
            b0[8 + j] = (_Float16)krow[16 + boff + j];
            b1[j]     = (_Float16)krow[32 + boff + j];
            b1[8 + j] = (_Float16)krow[48 + boff + j];
        }
        v8f acc = {};
        acc = __builtin_amdgcn_wmma_f32_16x16x32_f16(false, a0, false, b0,
                                                     (short)0, acc, false, false);
        acc = __builtin_amdgcn_wmma_f32_16x16x32_f16(false, a1, false, b1,
                                                     (short)0, acc, false, false);
        #pragma unroll
        for (int vr = 0; vr < 8; ++vr) {
            const int r = vr + half * 8;
            const int c = kt * 16 + l16;
            float s = acc[vr] * 0.125f;
            if (Mb[r * NK + c] == 0) s = -1e9f;
            Pw[r * 256 + (c ^ ((r & 7) << 3))] = s;
        }
    }

    // ---- softmax per row, write attention to global + normalized P to LDS ----
    for (int r = 0; r < 16; ++r) {
        const int sbase = r * 256 + ((lane * 8) ^ ((r & 7) << 3));
        float vals[8];
        float lm = -3.0e38f;
        #pragma unroll
        for (int i = 0; i < 8; ++i) { vals[i] = Pw[sbase + i]; lm = fmaxf(lm, vals[i]); }
        #pragma unroll
        for (int xm = 16; xm >= 1; xm >>= 1) lm = fmaxf(lm, __shfl_xor(lm, xm, 32));
        float ls = 0.f;
        #pragma unroll
        for (int i = 0; i < 8; ++i) { vals[i] = __expf(vals[i] - lm); ls += vals[i]; }
        #pragma unroll
        for (int xm = 16; xm >= 1; xm >>= 1) ls += __shfl_xor(ls, xm, 32);
        const float inv = 1.0f / ls;
        f32x4 o0, o1;
        #pragma unroll
        for (int i = 0; i < 8; ++i) {
            float p = vals[i] * inv;
            Pw[sbase + i] = p;
            if (i < 4) o0[i] = p; else o1[i - 4] = p;
        }
        f32x4* g = (f32x4*)(attb + (size_t)r * NK + lane * 8);
        g[0] = o0;
        g[1] = o1;
    }

    // ---- v_val = P @ V  -> out[..., 0:64] ----
    const int xr = (l16 & 7) << 3;       // A-row swizzle constant
    for (int nt = 0; nt < 4; ++nt) {
        v8f acc = {};
        #pragma unroll
        for (int kt = 0; kt < 8; ++kt) {
            const int kb = kt * 32;
            v16h a, bb;
            const float* prow = Pw + l16 * 256;
            #pragma unroll
            for (int j = 0; j < 8; ++j) {
                a[j]     = (_Float16)prow[((kb +      aoff) ^ xr) + j];
                a[8 + j] = (_Float16)prow[((kb + 16 + aoff) ^ xr) + j];
            }
            const float* vcol = Vb + (size_t)(kb + boff) * DK + nt * 16 + l16;
            #pragma unroll
            for (int j = 0; j < 8; ++j) { bb[j] = (_Float16)vcol[(size_t)j * DK]; bb[8 + j] = (_Float16)vcol[(size_t)(16 + j) * DK]; }
            acc = __builtin_amdgcn_wmma_f32_16x16x32_f16(false, a, false, bb,
                                                         (short)0, acc, false, false);
        }
        #pragma unroll
        for (int vr = 0; vr < 8; ++vr) {
            const int r = vr + half * 8;
            outb[(size_t)r * 128 + nt * 16 + l16] = acc[vr];
        }
    }
}

extern "C" void kernel_launch(void* const* d_in, const int* in_sizes, int n_in,
                              void* d_out, int out_size, void* d_ws, size_t ws_size,
                              hipStream_t stream) {
    (void)in_sizes; (void)n_in; (void)out_size; (void)d_ws; (void)ws_size;
    const float* q  = (const float*)d_in[0];
    const float* k  = (const float*)d_in[1];
    const float* v  = (const float*)d_in[2];
    const float* sa = (const float*)d_in[3];
    const float* sv = (const float*)d_in[4];
    const int*   m  = (const int*)d_in[5];
    float* out  = (float*)d_out;
    float* attn = out + (size_t)BSNH * NK * 128;

    attn_kernel<<<dim3(BSNH, 4), 128, 0, stream>>>(q, k, v, m, out, attn);
    sval_kernel<<<dim3(BSNH * NK), 256, 0, stream>>>(sa, sv, out);
}
